// DMPNNEncoderLayer_11158325035420
// MI455X (gfx1250) — matrix operand order repaired and checked by reading, not compiled, not run on hardware
//
#include <hip/hip_runtime.h>
#include <hip/hip_bf16.h>

// ---------------------------------------------------------------------------
// DMPNN encoder for MI455X (gfx1250, wave32).
//   - edge messages h kept in bf16 (halves dominant HBM traffic, feeds WMMA)
//   - node accumulator tmp f32, 25.6MB -> resident in 192MB L2 (gathers +
//     102M global_atomic_add_f32 hit L2, not HBM)
//   - GEMMs: D = W_tile x X_tile^T via v_wmma_f32_16x16x32_bf16 so each lane
//     ends up owning 8 consecutive output channels of one row -> single
//     packed 16B store per lane (no per-element exec-mask branches)
// ---------------------------------------------------------------------------

#define HIDDEN 128
#define AFDIM  133
#define BFDIM  14
#define KPAD_I 160   // 133+14=147 -> 5 K-chunks of 32
#define KPAD_H 128   // 4 K-chunks
#define KPAD_O 288   // 133+128=261 -> 9 K-chunks
#define RPB    32    // rows (edges/nodes) per block: two 16-wide B tiles

typedef __attribute__((ext_vector_type(8)))  float   v8f;
typedef __attribute__((ext_vector_type(8)))  __bf16  v8bf;
typedef __attribute__((ext_vector_type(16))) __bf16  v16bf;

union V16 { v16bf v; v8bf h[2]; };

__device__ __forceinline__ unsigned int f2bfbits(float f) {
  unsigned int u = __float_as_uint(f);
  u += 0x7FFFu + ((u >> 16) & 1u);          // round-to-nearest-even
  return u >> 16;
}
__device__ __forceinline__ float bf2f(unsigned short s) {
  return __uint_as_float(((unsigned int)s) << 16);
}

// A fragment = weight tile rows (16 output channels), 16-bit A 16x32
// operand, k order used here: lane m holds 16 CONTIGUOUS K kc+hi*16..+15 (hi = lane>=16).
// B fragment = staged row (edge/node) in LDS: lane n holds 16 CONTIGUOUS K
// values (lanes 0-15 -> K kc..kc+15, lanes 16-31 -> kc+16..kc+31).
// Two B tiles (rows i and i+16) share each A fragment.
template<int KPAD>
__device__ __forceinline__ void mma_pair(const unsigned short* __restrict__ Wb,
                                         const unsigned short* As,
                                         int lane, int w, v8f& acc0, v8f& acc1) {
  const int i  = lane & 15;
  const int hi = lane >> 4;
  const unsigned short* wrow  = Wb + (size_t)((w << 4) + i) * KPAD;  // channel row
  const unsigned short* xrow0 = As + i * KPAD + (hi << 4);
  const unsigned short* xrow1 = As + (i + 16) * KPAD + (hi << 4);
#pragma unroll
  for (int kc = 0; kc < KPAD; kc += 32) {
    V16 a;
    a.h[0] = *(const v8bf*)(wrow + kc + (hi << 4));        // global 16B
    a.h[1] = *(const v8bf*)(wrow + kc + (hi << 4) + 8);    // global 16B
    v16bf b0 = *(const v16bf*)(xrow0 + kc);                // 2x ds_load_b128
    v16bf b1 = *(const v16bf*)(xrow1 + kc);
    acc0 = __builtin_amdgcn_wmma_f32_16x16x32_bf16(false, a.v, false, b0,
                                                   (short)0, acc0, false, false);
    acc1 = __builtin_amdgcn_wmma_f32_16x16x32_bf16(false, a.v, false, b1,
                                                   (short)0, acc1, false, false);
  }
}

// relu + pack 8 f32 -> 8 bf16, single 16B store
__device__ __forceinline__ void store_bf16x8(unsigned short* dst, v8f a) {
  uint4 u;
  u.x = f2bfbits(fmaxf(a[0], 0.f)) | (f2bfbits(fmaxf(a[1], 0.f)) << 16);
  u.y = f2bfbits(fmaxf(a[2], 0.f)) | (f2bfbits(fmaxf(a[3], 0.f)) << 16);
  u.z = f2bfbits(fmaxf(a[4], 0.f)) | (f2bfbits(fmaxf(a[5], 0.f)) << 16);
  u.w = f2bfbits(fmaxf(a[6], 0.f)) | (f2bfbits(fmaxf(a[7], 0.f)) << 16);
  *(uint4*)dst = u;
}
__device__ __forceinline__ void store_f32x8(float* dst, v8f a) {
  float4 lo = make_float4(fmaxf(a[0],0.f), fmaxf(a[1],0.f), fmaxf(a[2],0.f), fmaxf(a[3],0.f));
  float4 hi = make_float4(fmaxf(a[4],0.f), fmaxf(a[5],0.f), fmaxf(a[6],0.f), fmaxf(a[7],0.f));
  ((float4*)dst)[0] = lo;
  ((float4*)dst)[1] = hi;
}

// ---------------- weight convert + pad (f32 -> bf16, K padded) --------------
__global__ void k_cvt_pad(const float* __restrict__ W, unsigned short* __restrict__ out,
                          int K, int KP) {
  int i = blockIdx.x * blockDim.x + threadIdx.x;
  if (i >= HIDDEN * KP) return;
  int r = i / KP, k = i - r * KP;
  out[i] = (k < K) ? (unsigned short)f2bfbits(W[(size_t)r * K + k]) : (unsigned short)0;
}

__global__ void k_zero(float* __restrict__ p, int n) {
  int i = blockIdx.x * blockDim.x + threadIdx.x;
  if (i < n) p[i] = 0.0f;
}

// ------------- scatter-add: tmp[dst[e]] += h[e]  (f32 atomics, L2-resident) -
__global__ void k_scatter(const unsigned short* __restrict__ h,
                          const int* __restrict__ dst,
                          float* __restrict__ tmp, int E) {
  int t = blockIdx.x * blockDim.x + threadIdx.x;   // 32 threads/edge, 4 cols each
  int e = t >> 5;
  if (e >= E) return;
  int c = (t & 31) << 2;
  int d = dst[e];
  uint2 packed = *(const uint2*)(h + (size_t)e * HIDDEN + c);   // 4 bf16, one b64 load
  float* tp = tmp + (size_t)d * HIDDEN + c;
  unsafeAtomicAdd(tp + 0, bf2f((unsigned short)(packed.x & 0xFFFF)));
  unsafeAtomicAdd(tp + 1, bf2f((unsigned short)(packed.x >> 16)));
  unsafeAtomicAdd(tp + 2, bf2f((unsigned short)(packed.y & 0xFFFF)));
  unsafeAtomicAdd(tp + 3, bf2f((unsigned short)(packed.y >> 16)));
}

// ---------------- init GEMM: h = relu([atom[src], bond] @ Wi^T) -------------
__global__ void __launch_bounds__(256)
k_init(const float* __restrict__ atom, const float* __restrict__ bond,
       const unsigned short* __restrict__ WiB, const int* __restrict__ src,
       unsigned short* __restrict__ h, int E) {
  __shared__ __align__(32) unsigned short As[RPB * KPAD_I];
  __shared__ int sSrc[RPB];
  const int tid = threadIdx.x;
  const int ebase = blockIdx.x * RPB;
  if (tid < RPB) { int e = ebase + tid; sSrc[tid] = (e < E) ? src[e] : 0; }
  __syncthreads();
  for (int idx = tid; idx < RPB * KPAD_I; idx += 256) {
    int r = idx / KPAD_I, k = idx - r * KPAD_I;
    int e = ebase + r;
    float v = 0.0f;
    if (e < E) {
      if (k < AFDIM)              v = atom[(size_t)sSrc[r] * AFDIM + k];
      else if (k < AFDIM + BFDIM) v = bond[(size_t)e * BFDIM + (k - AFDIM)];
    }
    As[idx] = (unsigned short)f2bfbits(v);
  }
  __syncthreads();
  const int lane = tid & 31, w = tid >> 5, hi = lane >> 4;
  v8f acc0 = {}, acc1 = {};
  mma_pair<KPAD_I>(WiB, As, lane, w, acc0, acc1);
  const int e0 = ebase + (lane & 15);
  const int ch = (w << 4) + (hi << 3);          // 8 consecutive channels
  if (e0 < E)      store_bf16x8(h + (size_t)e0 * HIDDEN + ch, acc0);
  if (e0 + 16 < E) store_bf16x8(h + (size_t)(e0 + 16) * HIDDEN + ch, acc1);
}

// --------- message GEMM: h' = relu((tmp[src] - hprev[rev]) @ Wh^T) ----------
__global__ void __launch_bounds__(256)
k_msg(const float* __restrict__ tmp, const unsigned short* __restrict__ hprev,
      const unsigned short* __restrict__ WhB, const int* __restrict__ src,
      const int* __restrict__ rev, unsigned short* __restrict__ hout, int E) {
  __shared__ __align__(32) unsigned short As[RPB * KPAD_H];
  __shared__ int sIdx[2 * RPB];
  const int tid = threadIdx.x;
  const int ebase = blockIdx.x * RPB;
  if (tid < RPB) {
    int e = ebase + tid;
    sIdx[tid]       = (e < E) ? src[e] : 0;
    sIdx[RPB + tid] = (e < E) ? rev[e] : 0;
  }
  __syncthreads();
  for (int idx = tid; idx < RPB * KPAD_H; idx += 256) {
    int r = idx >> 7, k = idx & (KPAD_H - 1);
    int e = ebase + r;
    float v = 0.0f;
    if (e < E)
      v = tmp[(size_t)sIdx[r] * HIDDEN + k]
        - bf2f(hprev[(size_t)sIdx[RPB + r] * HIDDEN + k]);
    As[idx] = (unsigned short)f2bfbits(v);
  }
  __syncthreads();
  const int lane = tid & 31, w = tid >> 5, hi = lane >> 4;
  v8f acc0 = {}, acc1 = {};
  mma_pair<KPAD_H>(WhB, As, lane, w, acc0, acc1);
  const int e0 = ebase + (lane & 15);
  const int ch = (w << 4) + (hi << 3);
  if (e0 < E)      store_bf16x8(hout + (size_t)e0 * HIDDEN + ch, acc0);
  if (e0 + 16 < E) store_bf16x8(hout + (size_t)(e0 + 16) * HIDDEN + ch, acc1);
}

// ---------- output GEMM: out = relu([atom, tmp] @ Wo^T) per node ------------
__global__ void __launch_bounds__(256)
k_out(const float* __restrict__ atom, const float* __restrict__ tmp,
      const unsigned short* __restrict__ WoB, float* __restrict__ out, int Nn) {
  __shared__ __align__(32) unsigned short As[RPB * KPAD_O];
  const int tid = threadIdx.x;
  const int nbase = blockIdx.x * RPB;
  for (int idx = tid; idx < RPB * KPAD_O; idx += 256) {
    int r = idx / KPAD_O, k = idx - r * KPAD_O;
    int n = nbase + r;
    float v = 0.0f;
    if (n < Nn) {
      if (k < AFDIM)               v = atom[(size_t)n * AFDIM + k];
      else if (k < AFDIM + HIDDEN) v = tmp[(size_t)n * HIDDEN + (k - AFDIM)];
    }
    As[idx] = (unsigned short)f2bfbits(v);
  }
  __syncthreads();
  const int lane = tid & 31, w = tid >> 5, hi = lane >> 4;
  v8f acc0 = {}, acc1 = {};
  mma_pair<KPAD_O>(WoB, As, lane, w, acc0, acc1);
  const int n0 = nbase + (lane & 15);
  const int ch = (w << 4) + (hi << 3);
  if (n0 < Nn)      store_f32x8(out + (size_t)n0 * HIDDEN + ch, acc0);
  if (n0 + 16 < Nn) store_f32x8(out + (size_t)(n0 + 16) * HIDDEN + ch, acc1);
}

// ---------------------------------------------------------------------------
extern "C" void kernel_launch(void* const* d_in, const int* in_sizes, int n_in,
                              void* d_out, int out_size, void* d_ws, size_t ws_size,
                              hipStream_t stream) {
  const float* atom = (const float*)d_in[0];
  const float* bond = (const float*)d_in[1];
  const float* Wi   = (const float*)d_in[2];
  const float* Wh   = (const float*)d_in[3];
  const float* Wo   = (const float*)d_in[4];
  const int*   src  = (const int*)d_in[5];
  const int*   dst  = (const int*)d_in[6];
  const int*   rev  = (const int*)d_in[7];

  const int Nn = in_sizes[0] / AFDIM;   // 50000
  const int E  = in_sizes[1] / BFDIM;   // 800000

  char* ws = (char*)d_ws;
  size_t off = 0;
  auto carve = [&](size_t bytes) -> char* {
    char* p = ws + off;
    off = (off + bytes + 255) & ~(size_t)255;
    return p;
  };
  unsigned short* WiB = (unsigned short*)carve((size_t)HIDDEN * KPAD_I * 2);
  unsigned short* WhB = (unsigned short*)carve((size_t)HIDDEN * KPAD_H * 2);
  unsigned short* WoB = (unsigned short*)carve((size_t)HIDDEN * KPAD_O * 2);
  unsigned short* hA  = (unsigned short*)carve((size_t)E * HIDDEN * 2);
  unsigned short* hB  = (unsigned short*)carve((size_t)E * HIDDEN * 2);
  float*          tmp = (float*)carve((size_t)Nn * HIDDEN * 4);
  if (off > ws_size) return;  // workspace too small: deterministic no-op

  const int ntmp   = Nn * HIDDEN;
  const int etiles = (E + RPB - 1) / RPB;
  const int ntiles = (Nn + RPB - 1) / RPB;

  k_cvt_pad<<<(HIDDEN * KPAD_I + 255) / 256, 256, 0, stream>>>(Wi, WiB, AFDIM + BFDIM, KPAD_I);
  k_cvt_pad<<<(HIDDEN * KPAD_H + 255) / 256, 256, 0, stream>>>(Wh, WhB, HIDDEN, KPAD_H);
  k_cvt_pad<<<(HIDDEN * KPAD_O + 255) / 256, 256, 0, stream>>>(Wo, WoB, AFDIM + HIDDEN, KPAD_O);

  k_init<<<etiles, 256, 0, stream>>>(atom, bond, WiB, src, hA, E);

  unsigned short* hcur = hA;
  unsigned short* hnxt = hB;
  for (int it = 0; it < 2; ++it) {  // DEPTH-1
    k_zero<<<(ntmp + 255) / 256, 256, 0, stream>>>(tmp, ntmp);
    k_scatter<<<(int)(((size_t)E * 32 + 255) / 256), 256, 0, stream>>>(hcur, dst, tmp, E);
    k_msg<<<etiles, 256, 0, stream>>>(tmp, hcur, WhB, src, rev, hnxt, E);
    unsigned short* t = hcur; hcur = hnxt; hnxt = t;
  }

  k_zero<<<(ntmp + 255) / 256, 256, 0, stream>>>(tmp, ntmp);
  k_scatter<<<(int)(((size_t)E * 32 + 255) / 256), 256, 0, stream>>>(hcur, dst, tmp, E);
  k_out<<<ntiles, 256, 0, stream>>>(atom, tmp, WoB, (float*)d_out, Nn);
}
